// Supervised_NSM_Conv_25297357373650
// MI455X (gfx1250) — matrix operand order repaired and checked by reading, not compiled, not run on hardware
//
#include <hip/hip_runtime.h>

typedef __attribute__((ext_vector_type(16))) _Float16 v16h;
typedef __attribute__((ext_vector_type(8)))  float    v8f;
typedef __attribute__((ext_vector_type(8)))  unsigned v8u;

// ---------------- geometry ----------------
constexpr int BATCH = 32, CIN = 3, CLAB = 10, HIN = 64, WIN = 64, KK = 8;
constexpr int OH = 57, OW = 57, PPI = OH * OW;          // 3249
constexpr int TOTPIX = BATCH * PPI;                     // 103968 = 32*3249
constexpr int C = 128;
constexpr int QX = CIN * KK * KK;                       // 192
constexpr int QTOT = QX + CLAB * KK * KK;               // 832
constexpr int KC_CONV = QTOT / 32;                      // 26
constexpr int NBB = 8;                                  // 8 row-blocks of 16 channels
constexpr int NKC = 4;                                  // 128/32 K-chunks for M
constexpr int NT  = 2;                                  // 2 B-tiles (32 pixels) per wave
constexpr int PIX_PER_WAVE = 16 * NT;                   // 32
constexpr int NITER = 40;

constexpr int WAVES_PER_WG = 4;
constexpr int THREADS = WAVES_PER_WG * 32;

// ---------------- workspace layout ----------------
constexpr size_t WSW_HALFS = (size_t)NBB * KC_CONV * 32 * 16;   // 106496
constexpr size_t WSM_HALFS = (size_t)NBB * NKC * 32 * 16;       // 16384
constexpr size_t WSW_OFF = 0;
constexpr size_t WSM_OFF = WSW_HALFS * 2;
constexpr size_t BIAS_OFF = WSM_OFF + WSM_HALFS * 2;

__device__ __forceinline__ unsigned pkf16(float a, float b) {
  union { _Float16 h; unsigned short u; } ua, ub;
  ua.h = (_Float16)a; ub.h = (_Float16)b;
  return (unsigned)ua.u | ((unsigned)ub.u << 16);
}

__device__ __forceinline__ v8f wmma16(v16h a, v16h b, v8f c) {
  return __builtin_amdgcn_wmma_f32_16x16x32_f16(false, a, false, b, (short)0, c, false, false);
}

// ---------------- one-time pre-pack ----------------
// Conv weights: A position Kloc within chunk kc holds q = 32*kc + Kloc (identity K-order,
// matches the gathered B built as position Kloc = 16*hi + h -> q = 32*kc + Kloc).
// -M: K-order permuted so B can be packed lane-locally from the C-layout of y:
//   position Kloc -> j = 16*(2*kc + ((Kloc&15)>=8)) + (Kloc&7) + 8*(Kloc>>4)
__global__ void nsm_prepack(const float* __restrict__ W,  const float* __restrict__ Wl,
                            const float* __restrict__ Mw, const float* __restrict__ b,
                            const float* __restrict__ bl,
                            unsigned short* __restrict__ wsW,
                            unsigned short* __restrict__ wsM,
                            float* __restrict__ biasSum) {
  int idx = blockIdx.x * blockDim.x + threadIdx.x;
  if (idx < (int)WSW_HALFS) {
    int h = idx & 15, lane = (idx >> 4) & 31;
    int t = idx >> 9; int kc = t % KC_CONV, bb = t / KC_CONV;
    int c = 16 * bb + (lane & 15), hi = lane >> 4;
    int kloc = h + 16 * hi;                          // contiguous K pattern (matches gathered B)
    int q = 32 * kc + kloc;                          // < 832 always
    float v = (q < QX) ? W[(size_t)c * QX + q] : Wl[(size_t)c * (QTOT - QX) + (q - QX)];
    _Float16 hv = (_Float16)v;
    wsW[idx] = __builtin_bit_cast(unsigned short, hv);
  }
  int i2 = idx - (int)WSW_HALFS;
  if (i2 >= 0 && i2 < (int)WSM_HALFS) {
    int h = i2 & 15, lane = (i2 >> 4) & 31;
    int t = i2 >> 9; int kc = t & 3, ib = t >> 2;
    int i = 16 * ib + (lane & 15), hi = lane >> 4;
    int kloc = (h < 8 ? h : h + 8) + 8 * hi;
    int j = 32 * kc + kloc;
    float m = Mw[(size_t)i * C + j] - (i == j ? 1.0f : 0.0f);
    _Float16 hv = (_Float16)(-m);
    wsM[i2] = __builtin_bit_cast(unsigned short, hv);
  }
  int i3 = idx - (int)(WSW_HALFS + WSM_HALFS);
  if (i3 >= 0 && i3 < C) biasSum[i3] = b[i3] + bl[i3];
}

// ---------------- main kernel ----------------
__global__ __launch_bounds__(THREADS) void nsm_iterate(
    const float* __restrict__ x, const float* __restrict__ label,
    const unsigned short* __restrict__ wsW, const unsigned short* __restrict__ wsM,
    const float* __restrict__ biasSum, float* __restrict__ out) {

  __shared__ __align__(16) unsigned short sM[WSM_HALFS];   // 32 KB -M tiles
  {
    const uint4* src = (const uint4*)wsM;
    uint4* dst = (uint4*)sM;
    for (int i = threadIdx.x; i < (int)(WSM_HALFS / 8); i += THREADS) dst[i] = src[i];
  }
  __syncthreads();

  const int lane = threadIdx.x & 31;
  const int wave = threadIdx.x >> 5;
  const int hi   = lane >> 4;
  const int col  = lane & 15;

  long tile = (long)blockIdx.x * WAVES_PER_WG + wave;

  bool   valid[NT];
  size_t outBase[NT];
  const float* xb[NT];
  const float* lb[NT];
  #pragma unroll
  for (int t = 0; t < NT; ++t) {
    long pp = tile * PIX_PER_WAVE + 16 * t + col;
    bool v = pp < (long)TOTPIX;
    long p = v ? pp : (long)(TOTPIX - 1);            // clamp: EXEC stays all-ones
    int bimg = (int)(p / PPI);
    int rem  = (int)(p % PPI);
    int oh = rem / OW, ow = rem % OW;
    valid[t] = v;
    xb[t] = x     + ((size_t)bimg * CIN)  * (HIN * WIN) + (size_t)oh * WIN + ow;
    lb[t] = label + ((size_t)bimg * CLAB) * (HIN * WIN) + (size_t)oh * WIN + ow;
    outBase[t] = (size_t)bimg * C * PPI + (size_t)oh * OW + ow;
  }

  // ---- Phase 1: drive = conv(x,W) + conv(label,Wl), im2col GEMM with K=832 ----
  v8f drive[NT][NBB];
  #pragma unroll
  for (int t = 0; t < NT; ++t)
    #pragma unroll
    for (int bb = 0; bb < NBB; ++bb)
      #pragma unroll
      for (int r = 0; r < 8; ++r) drive[t][bb][r] = 0.0f;

  const v16h* wtiles = (const v16h*)wsW;
  for (int kc = 0; kc < KC_CONV; ++kc) {
    v16h bvec[NT];
    #pragma unroll
    for (int t = 0; t < NT; ++t) {
      v8u bu;
      #pragma unroll
      for (int hp = 0; hp < 8; ++hp) {
        float f[2];
        #pragma unroll
        for (int e = 0; e < 2; ++e) {
          int q = 32 * kc + 16 * hi + 2 * hp + e;
          int qq = (q < QX) ? q : (q - QX);
          const float* base = (q < QX) ? xb[t] : lb[t];
          int cin = qq >> 6, rr = qq & 63;
          f[e] = base[(size_t)cin * (HIN * WIN) + (rr >> 3) * WIN + (rr & 7)];
        }
        bu[hp] = pkf16(f[0], f[1]);
      }
      bvec[t] = __builtin_bit_cast(v16h, bu);
    }
    #pragma unroll
    for (int bb = 0; bb < NBB; ++bb) {
      v16h a = wtiles[(size_t)(bb * KC_CONV + kc) * 32 + lane];
      #pragma unroll
      for (int t = 0; t < NT; ++t)
        drive[t][bb] = wmma16(a, bvec[t], drive[t][bb]);
    }
  }
  #pragma unroll
  for (int bb = 0; bb < NBB; ++bb) {
    #pragma unroll
    for (int r = 0; r < 8; ++r) {
      float bs = biasSum[16 * bb + r + 8 * hi];
      #pragma unroll
      for (int t = 0; t < NT; ++t) drive[t][bb][r] += bs;
    }
  }

  // ---- iteration 1: y = relu(drive), packed lane-locally (permuted K-order) ----
  v16h ybA[NT][NKC], ybB[NT][NKC];
  #pragma unroll
  for (int kc2 = 0; kc2 < NKC; ++kc2)
    #pragma unroll
    for (int t = 0; t < NT; ++t) {
      v8u u;
      #pragma unroll
      for (int r = 0; r < 4; ++r) {
        u[r]     = pkf16(fmaxf(drive[t][2 * kc2][2 * r], 0.f),
                         fmaxf(drive[t][2 * kc2][2 * r + 1], 0.f));
        u[4 + r] = pkf16(fmaxf(drive[t][2 * kc2 + 1][2 * r], 0.f),
                         fmaxf(drive[t][2 * kc2 + 1][2 * r + 1], 0.f));
      }
      ybA[t][kc2] = __builtin_bit_cast(v16h, u);
    }

  const v16h* mt = (const v16h*)sM;

  // one fixed-point step: yout = pack(relu(drive + (-M) * yin))
  auto step = [&](v16h (&yin)[NT][NKC], v16h (&yout)[NT][NKC]) {
    #pragma unroll
    for (int kc2 = 0; kc2 < NKC; ++kc2) {
      const int bbE = 2 * kc2, bbO = bbE + 1;
      v8f accE[NT], accO[NT];
      #pragma unroll
      for (int t = 0; t < NT; ++t) { accE[t] = drive[t][bbE]; accO[t] = drive[t][bbO]; }
      #pragma unroll
      for (int kc = 0; kc < NKC; ++kc) {
        v16h ae = mt[(size_t)(bbE * NKC + kc) * 32 + lane];
        v16h ao = mt[(size_t)(bbO * NKC + kc) * 32 + lane];
        #pragma unroll
        for (int t = 0; t < NT; ++t) accE[t] = wmma16(ae, yin[t][kc], accE[t]);
        #pragma unroll
        for (int t = 0; t < NT; ++t) accO[t] = wmma16(ao, yin[t][kc], accO[t]);
      }
      #pragma unroll
      for (int t = 0; t < NT; ++t) {
        v8u u;
        #pragma unroll
        for (int r = 0; r < 4; ++r) {
          u[r]     = pkf16(fmaxf(accE[t][2 * r], 0.f), fmaxf(accE[t][2 * r + 1], 0.f));
          u[4 + r] = pkf16(fmaxf(accO[t][2 * r], 0.f), fmaxf(accO[t][2 * r + 1], 0.f));
        }
        yout[t][kc2] = __builtin_bit_cast(v16h, u);
      }
    }
  };

  // ---- iterations 2..39: 19 double-steps (A->B->A), y stays in buffer A ----
  for (int k = 0; k < (NITER - 2) / 2; ++k) {   // 19
    step(ybA, ybB);
    step(ybB, ybA);
  }

  // ---- iteration 40 (peeled): compute, relu, store directly ----
  #pragma unroll
  for (int kc2 = 0; kc2 < NKC; ++kc2) {
    const int bbE = 2 * kc2, bbO = bbE + 1;
    v8f accE[NT], accO[NT];
    #pragma unroll
    for (int t = 0; t < NT; ++t) { accE[t] = drive[t][bbE]; accO[t] = drive[t][bbO]; }
    #pragma unroll
    for (int kc = 0; kc < NKC; ++kc) {
      v16h ae = mt[(size_t)(bbE * NKC + kc) * 32 + lane];
      v16h ao = mt[(size_t)(bbO * NKC + kc) * 32 + lane];
      #pragma unroll
      for (int t = 0; t < NT; ++t) accE[t] = wmma16(ae, ybA[t][kc], accE[t]);
      #pragma unroll
      for (int t = 0; t < NT; ++t) accO[t] = wmma16(ao, ybA[t][kc], accO[t]);
    }
    #pragma unroll
    for (int t = 0; t < NT; ++t) {
      if (valid[t]) {
        #pragma unroll
        for (int r = 0; r < 8; ++r) {
          out[outBase[t] + (size_t)(16 * bbE + r + 8 * hi) * PPI] = fmaxf(accE[t][r], 0.f);
          out[outBase[t] + (size_t)(16 * bbO + r + 8 * hi) * PPI] = fmaxf(accO[t][r], 0.f);
        }
      }
    }
  }
}

// ---------------- launch ----------------
extern "C" void kernel_launch(void* const* d_in, const int* in_sizes, int n_in,
                              void* d_out, int out_size, void* d_ws, size_t ws_size,
                              hipStream_t stream) {
  const float* x     = (const float*)d_in[0];
  const float* label = (const float*)d_in[1];
  const float* W     = (const float*)d_in[2];
  const float* b     = (const float*)d_in[3];
  const float* Wl    = (const float*)d_in[4];
  const float* bl    = (const float*)d_in[5];
  const float* Mw    = (const float*)d_in[6];

  unsigned short* wsW = (unsigned short*)((char*)d_ws + WSW_OFF);
  unsigned short* wsM = (unsigned short*)((char*)d_ws + WSM_OFF);
  float* biasSum      = (float*)((char*)d_ws + BIAS_OFF);

  int nPre = (int)(WSW_HALFS + WSM_HALFS + C);
  nsm_prepack<<<(nPre + 255) / 256, 256, 0, stream>>>(W, Wl, Mw, b, bl, wsW, wsM, biasSum);

  int tiles = (TOTPIX + PIX_PER_WAVE - 1) / PIX_PER_WAVE;      // 3249
  int wgs = (tiles + WAVES_PER_WG - 1) / WAVES_PER_WG;         // 813
  nsm_iterate<<<wgs, THREADS, 0, stream>>>(x, label, wsW, wsM, biasSum, (float*)d_out);
}
